// AdaptiveInterventionEngine_47399259078749
// MI455X (gfx1250) — compile-verified, matrix operand order checked by reading, not run on hardware
//
#include <hip/hip_runtime.h>
#include <hip/hip_bf16.h>
#include <math.h>

// ---------------------------------------------------------------------------
// MI455X (gfx1250, wave32). All GEMMs via v_wmma_f32_16x16x32_bf16
// (bf16 inputs, f32 accumulate). B tiles staged TRANSPOSED in LDS so both
// A and B fragments load as two contiguous ds_load_b128 per lane.
// Each wave owns a 32x64 output tile (2 M-tiles x 4 N-tiles) so every B
// fragment feeds two WMMAs (1.5 ds_load_b128 per WMMA).
// ---------------------------------------------------------------------------

typedef __bf16 bf16t;
typedef bf16t v16bf __attribute__((ext_vector_type(16)));
typedef bf16t v8bf  __attribute__((ext_vector_type(8)));
typedef bf16t v4bf  __attribute__((ext_vector_type(4)));
typedef float v8f   __attribute__((ext_vector_type(8)));

static constexpr int kBatch  = 32768;
static constexpr int kDC     = 512;    // d_causal
static constexpr int kDA     = 1024;   // d_action
static constexpr int kDH     = 2048;   // hidden
static constexpr int kEnc    = 128;    // ENC_D
static constexpr int kKHalt  = kDC + kEnc;  // 640
static constexpr int kMaxDepth = 6;

static constexpr int kALd = 72;   // LDS leading dim, A tile (128x64), padded
static constexpr int kBLd = 72;   // LDS leading dim, transposed B tile (128x64)
static constexpr int kHLd = 40;   // LDS leading dim in halter (K-step 32), padded

__device__ __forceinline__ float gelu_exact(float x) {
    return 0.5f * x * (1.0f + erff(x * 0.70710678118654752f));
}
__device__ __forceinline__ float sigmoid_(float x) {
    return 1.0f / (1.0f + expf(-x));
}

// 16-bit fragment loader (CDNA5 ISA 7.12.2): lane (half,r), element j<8 ->
// K=half*8+j ; j>=8 -> K=16+half*8+(j-8). Rows contiguous in LDS => 2x b128.
// Used for A tiles and for TRANSPOSED B tiles (N plays the row role).
__device__ __forceinline__ v16bf frag16_lds(const bf16t* base, int ld, int lane) {
    const int half = lane >> 4, r = lane & 15;
    const bf16t* p = base + r * ld + half * 8;
    v16bf f;
#pragma unroll
    for (int j = 0; j < 8; ++j) { f[j] = p[j]; f[j + 8] = p[j + 16]; }
    return f;
}

// ------------ 128(M) x 128(N) block tile GEMM mainloop, 256 threads ---------
// K-step 64. wave w: m-base (w&3)*32 (2 tiles), n-base (w>>2)*64 (4 tiles).
__device__ __forceinline__ void gemm_tile_128x128(
    const bf16t* __restrict__ A, int lda,
    const bf16t* __restrict__ Bm, int ldb,
    int K, int mBlock, int nBlock,
    bf16t* aT, bf16t* bTt, v8f acc[2][4])
{
    const int tid = threadIdx.x;
    const int lane = tid & 31, w = tid >> 5;
    const int mt0 = (w & 3) * 32;
    const int nb  = (w >> 2) * 64;
#pragma unroll
    for (int mi = 0; mi < 2; ++mi)
#pragma unroll
        for (int t = 0; t < 4; ++t) acc[mi][t] = (v8f){0, 0, 0, 0, 0, 0, 0, 0};

    const int arow = tid >> 1;           // 0..127
    const int aseg = (tid & 1) * 32;     // 0, 32
    const int kq   = tid >> 5;           // 0..7  -> k block of 8
    const int nq   = tid & 31;           // 0..31 -> n block of 4
    const int k0l  = kq * 8, n0 = nq * 4;

    for (int k0 = 0; k0 < K; k0 += 64) {
        // ---- stage A tile 128x64 row-major (4x16B per thread) ----
        const bf16t* asrc = A + (size_t)(mBlock + arow) * lda + k0 + aseg;
#pragma unroll
        for (int u = 0; u < 4; ++u)
            *(uint4*)(aT + arow * kALd + aseg + u * 8) = *(const uint4*)(asrc + u * 8);
        if (k0 + 64 < K) __builtin_prefetch(asrc + 64, 0, 1);

        // ---- stage B tile 64x128 TRANSPOSED -> bTt[n][k] ----
        v4bf rows[8];
#pragma unroll
        for (int j = 0; j < 8; ++j)
            rows[j] = *(const v4bf*)(Bm + (size_t)(k0 + k0l + j) * ldb + nBlock + n0);
#pragma unroll
        for (int c = 0; c < 4; ++c) {
            v8bf col;
#pragma unroll
            for (int j = 0; j < 8; ++j) col[j] = rows[j][c];
            *(v8bf*)(bTt + (n0 + c) * kBLd + k0l) = col;
        }
        __syncthreads();

#pragma unroll
        for (int kc = 0; kc < 64; kc += 32) {
            v16bf a0 = frag16_lds(aT + (mt0 +  0) * kALd + kc, kALd, lane);
            v16bf a1 = frag16_lds(aT + (mt0 + 16) * kALd + kc, kALd, lane);
#pragma unroll
            for (int t = 0; t < 4; ++t) {
                v16bf b = frag16_lds(bTt + (nb + t * 16) * kBLd + kc, kBLd, lane);
                acc[0][t] = __builtin_amdgcn_wmma_f32_16x16x32_bf16(
                    false, a0, false, b, (short)0, acc[0][t], false, false);
                acc[1][t] = __builtin_amdgcn_wmma_f32_16x16x32_bf16(
                    false, a1, false, b, (short)0, acc[1][t], false, false);
            }
        }
        __syncthreads();
    }
}

// ------------------------------ kernels -------------------------------------

__global__ void k_f32_to_bf16(const float* __restrict__ src, bf16t* __restrict__ dst,
                              long long n4) {   // n4 = elements/4
    long long i = (long long)blockIdx.x * blockDim.x + threadIdx.x;
    if (i < n4) {
        float4 v = ((const float4*)src)[i];
        v4bf o;
        o[0] = (bf16t)v.x; o[1] = (bf16t)v.y; o[2] = (bf16t)v.z; o[3] = (bf16t)v.w;
        ((v4bf*)dst)[i] = o;
    }
}

__global__ void k_init_state(int* __restrict__ flags, int* __restrict__ steps) {
    if (threadIdx.x == 0) {
        for (int i = 0; i <= kMaxDepth; ++i) flags[i] = 0;
        steps[0] = 0;
    }
}

// y = A @ B + bias  (mode 0: plain fp32 out; mode 1: sigmoid fp32 out)
__global__ void k_gemm_bias(const bf16t* __restrict__ A, int lda,
                            const bf16t* __restrict__ Bm, int ldb,
                            int K, int N, const float* __restrict__ bias,
                            float* __restrict__ out, int mode) {
    __shared__ __align__(16) bf16t aT[128 * kALd];
    __shared__ __align__(16) bf16t bTt[128 * kBLd];
    v8f acc[2][4];
    const int mBlock = blockIdx.y * 128, nBlock = blockIdx.x * 128;
    gemm_tile_128x128(A, lda, Bm, ldb, K, mBlock, nBlock, aT, bTt, acc);

    const int lane = threadIdx.x & 31, w = threadIdx.x >> 5;
    const int mt0 = (w & 3) * 32, nb = (w >> 2) * 64;
    const int half = lane >> 4, r = lane & 15;
#pragma unroll
    for (int mi = 0; mi < 2; ++mi) {
#pragma unroll
        for (int t = 0; t < 4; ++t) {
            const int col = nBlock + nb + t * 16 + r;
            const float bi = bias[col];
#pragma unroll
            for (int i = 0; i < 8; ++i) {
                const int row = mBlock + mt0 + mi * 16 + half * 8 + i;
                float v = acc[mi][t][i] + bi;
                if (mode == 1) v = sigmoid_(v);
                out[(size_t)row * N + col] = v;
            }
        }
    }
}

// LayerNorm + GELU + cast to bf16, one row (length n) per block.
__global__ void k_ln_gelu_bf16(const float* __restrict__ x,
                               const float* __restrict__ g,
                               const float* __restrict__ b,
                               bf16t* __restrict__ out, int n) {
    __shared__ float red[256];
    const int row = blockIdx.x;
    const float* xr = x + (size_t)row * n;
    float s = 0.f, s2 = 0.f;
    for (int j = threadIdx.x; j < n; j += 256) { float v = xr[j]; s += v; s2 += v * v; }
    red[threadIdx.x] = s; __syncthreads();
    for (int o = 128; o > 0; o >>= 1) {
        if (threadIdx.x < o) red[threadIdx.x] += red[threadIdx.x + o];
        __syncthreads();
    }
    const float mean = red[0] / (float)n; __syncthreads();
    red[threadIdx.x] = s2; __syncthreads();
    for (int o = 128; o > 0; o >>= 1) {
        if (threadIdx.x < o) red[threadIdx.x] += red[threadIdx.x + o];
        __syncthreads();
    }
    const float var = red[0] / (float)n - mean * mean;
    const float rstd = rsqrtf(var + 1e-5f);
    bf16t* orow = out + (size_t)row * n;
    for (int j = threadIdx.x; j < n; j += 256) {
        float v = (xr[j] - mean) * rstd * g[j] + b[j];
        orow[j] = (bf16t)gelu_exact(v);
    }
}

// epsilon = z - z@adj ; z0 = mask*values + (1-mask)*z ; base = mask*z0
__global__ void k_gemm_eps(const bf16t* __restrict__ zb, const bf16t* __restrict__ adjb,
                           const float* __restrict__ z, const float* __restrict__ mask,
                           const float* __restrict__ values,
                           float* __restrict__ epsOut, float* __restrict__ baseOut,
                           float* __restrict__ zOut, bf16t* __restrict__ zbOut) {
    __shared__ __align__(16) bf16t aT[128 * kALd];
    __shared__ __align__(16) bf16t bTt[128 * kBLd];
    v8f acc[2][4];
    const int mBlock = blockIdx.y * 128, nBlock = blockIdx.x * 128;
    gemm_tile_128x128(zb, kDC, adjb, kDC, kDC, mBlock, nBlock, aT, bTt, acc);

    const int lane = threadIdx.x & 31, w = threadIdx.x >> 5;
    const int mt0 = (w & 3) * 32, nb = (w >> 2) * 64;
    const int half = lane >> 4, r = lane & 15;
#pragma unroll
    for (int mi = 0; mi < 2; ++mi) {
#pragma unroll
        for (int t = 0; t < 4; ++t) {
            const int col = nBlock + nb + t * 16 + r;
#pragma unroll
            for (int i = 0; i < 8; ++i) {
                const int row = mBlock + mt0 + mi * 16 + half * 8 + i;
                const size_t idx = (size_t)row * kDC + col;
                const float za = acc[mi][t][i];    // (z @ adjacency)[row,col]
                const float zz = z[idx];
                const float m  = mask[idx];
                const float z0 = m * values[idx] + (1.f - m) * zz;
                epsOut[idx]  = zz - za;
                baseOut[idx] = m * z0;
                zOut[idx]    = z0;
                zbOut[idx]   = (bf16t)z0;
            }
        }
    }
}

// one fixed-point iteration: z_new = base + (1-mask)*(z@adj + eps)
__global__ void k_gemm_iter(const bf16t* __restrict__ zbIn, const bf16t* __restrict__ adjb,
                            const float* __restrict__ zprev, const float* __restrict__ base,
                            const float* __restrict__ mask, const float* __restrict__ eps,
                            const int* __restrict__ flags, int s,
                            float* __restrict__ zOut, bf16t* __restrict__ zbOut,
                            bf16t* __restrict__ deltaOut) {
    __shared__ __align__(16) bf16t aT[128 * kALd];
    __shared__ __align__(16) bf16t bTt[128 * kBLd];
    v8f acc[2][4];
    const int mBlock = blockIdx.y * 128, nBlock = blockIdx.x * 128;
    gemm_tile_128x128(zbIn, kDC, adjb, kDC, kDC, mBlock, nBlock, aT, bTt, acc);

    const int done = flags[s];
    const int lane = threadIdx.x & 31, w = threadIdx.x >> 5;
    const int mt0 = (w & 3) * 32, nb = (w >> 2) * 64;
    const int half = lane >> 4, r = lane & 15;
#pragma unroll
    for (int mi = 0; mi < 2; ++mi) {
#pragma unroll
        for (int t = 0; t < 4; ++t) {
            const int col = nBlock + nb + t * 16 + r;
#pragma unroll
            for (int i = 0; i < 8; ++i) {
                const int row = mBlock + mt0 + mi * 16 + half * 8 + i;
                const size_t idx = (size_t)row * kDC + col;
                const float zp = zprev[idx];
                const float zn = base[idx] + (1.f - mask[idx]) * (acc[mi][t][i] + eps[idx]);
                const float zo = done ? zp : zn;
                zOut[idx]     = zo;
                zbOut[idx]    = (bf16t)zo;
                deltaOut[idx] = (bf16t)fabsf(zn - zp);
            }
        }
    }
}

// halter: hg = gelu([delta|enc] @ wh1 + bh1); hp = sigmoid(hg.wh2 + bh2)
// 32 rows per block; full N=512 in-block; K-step 32; wh1 transposed in LDS.
__global__ void k_halter(const bf16t* __restrict__ delta, const bf16t* __restrict__ wh1,
                         const float* __restrict__ bh1, const float* __restrict__ wh2,
                         const float* __restrict__ bh2, float* __restrict__ hpPart,
                         int step) {
    __shared__ __align__(16) bf16t bTt[512 * kHLd];   // 40 KB, transposed wh1 tile
    __shared__ __align__(16) bf16t aT[32 * kHLd];
    __shared__ bf16t enc[kEnc];
    __shared__ float partW[8][32];
    __shared__ float red[256];

    const int tid = threadIdx.x, lane = tid & 31, w = tid >> 5;
    const int mBlock = blockIdx.x * 32;

    if (tid < 64) {
        const float div = expf((float)(2 * tid) * (-9.210340371976184f / 128.0f));
        const float ang = (float)step * div;
        enc[2 * tid]     = (bf16t)sinf(ang);
        enc[2 * tid + 1] = (bf16t)cosf(ang);
    }
    __syncthreads();

    v8f acc[2][4];
#pragma unroll
    for (int mi = 0; mi < 2; ++mi)
#pragma unroll
        for (int t = 0; t < 4; ++t) acc[mi][t] = (v8f){0, 0, 0, 0, 0, 0, 0, 0};

    const int kq = tid >> 5;        // 0..7 -> 4 k-rows each
    const int nq = tid & 31;        // 0..31 -> 16 n-cols each
    const int n0 = nq * 16;

    for (int k0 = 0; k0 < kKHalt; k0 += 32) {
        {   // A tile 32x32: delta cols for k<512, broadcast enc for k>=512
            const int e0 = tid * 4;
#pragma unroll
            for (int u = 0; u < 4; ++u) {
                const int e = e0 + u, rr = e >> 5, kk = e & 31, k = k0 + kk;
                aT[rr * kHLd + kk] = (k < kDC)
                    ? delta[(size_t)(mBlock + rr) * kDC + k]
                    : enc[k - kDC];
            }
        }
        {   // B tile 32x512 TRANSPOSED -> bTt[n][k]
            v16bf rows[4];
#pragma unroll
            for (int j = 0; j < 4; ++j)
                rows[j] = *(const v16bf*)(wh1 + (size_t)(k0 + kq * 4 + j) * kDC + n0);
#pragma unroll
            for (int c = 0; c < 16; ++c) {
                v4bf col;
#pragma unroll
                for (int j = 0; j < 4; ++j) col[j] = rows[j][c];
                *(v4bf*)(bTt + (n0 + c) * kHLd + kq * 4) = col;
            }
        }
        __syncthreads();
        v16bf a0 = frag16_lds(aT,             kHLd, lane);
        v16bf a1 = frag16_lds(aT + 16 * kHLd, kHLd, lane);
#pragma unroll
        for (int t = 0; t < 4; ++t) {
            v16bf b = frag16_lds(bTt + (w * 64 + t * 16) * kHLd, kHLd, lane);
            acc[0][t] = __builtin_amdgcn_wmma_f32_16x16x32_bf16(
                false, a0, false, b, (short)0, acc[0][t], false, false);
            acc[1][t] = __builtin_amdgcn_wmma_f32_16x16x32_bf16(
                false, a1, false, b, (short)0, acc[1][t], false, false);
        }
        __syncthreads();
    }

    // fused epilogue: gelu(+bh1) dotted with wh2, reduced deterministically
    const int half = lane >> 4, r = lane & 15;
    float part[2][8];
#pragma unroll
    for (int mi = 0; mi < 2; ++mi)
#pragma unroll
        for (int i = 0; i < 8; ++i) part[mi][i] = 0.f;
#pragma unroll
    for (int t = 0; t < 4; ++t) {
        const int col = w * 64 + t * 16 + r;
        const float w2 = wh2[col];
        const float bb = bh1[col];
#pragma unroll
        for (int mi = 0; mi < 2; ++mi)
#pragma unroll
            for (int i = 0; i < 8; ++i)
                part[mi][i] += gelu_exact(acc[mi][t][i] + bb) * w2;
    }
#pragma unroll
    for (int mi = 0; mi < 2; ++mi)
#pragma unroll
        for (int i = 0; i < 8; ++i) {           // butterfly over the 16-lane group
            float v = part[mi][i];
            for (int m = 8; m >= 1; m >>= 1) v += __shfl_xor(v, m, 32);
            if (r == 0) partW[w][mi * 16 + half * 8 + i] = v;
        }
    __syncthreads();
    float contrib = 0.f;
    if (tid < 32) {
        float s = bh2[0];
        for (int ww = 0; ww < 8; ++ww) s += partW[ww][tid];
        contrib = sigmoid_(s);
    }
    red[tid] = contrib; __syncthreads();
    for (int o = 128; o > 0; o >>= 1) {
        if (tid < o) red[tid] += red[tid + o];
        __syncthreads();
    }
    if (tid == 0) hpPart[blockIdx.x] = red[0];
}

// deterministic final reduce + halt / done / steps bookkeeping
__global__ void k_halt_reduce(const float* __restrict__ hpPart, int nPart,
                              int* __restrict__ flags, int* __restrict__ steps,
                              int s, int* __restrict__ stepsOut) {
    __shared__ float red[256];
    float a = 0.f;
    for (int j = threadIdx.x; j < nPart; j += 256) a += hpPart[j];
    red[threadIdx.x] = a; __syncthreads();
    for (int o = 128; o > 0; o >>= 1) {
        if (threadIdx.x < o) red[threadIdx.x] += red[threadIdx.x + o];
        __syncthreads();
    }
    if (threadIdx.x == 0) {
        const float mean = red[0] / (float)kBatch;
        const int done = flags[s];
        const int halt = (mean > 0.5f) ? 1 : 0;
        steps[0] += done ? 0 : 1;
        flags[s + 1] = done | halt;
        if (s == kMaxDepth - 1) stepsOut[0] = steps[0];
    }
}

// ---------------------------------------------------------------------------

extern "C" void kernel_launch(void* const* d_in, const int* in_sizes, int n_in,
                              void* d_out, int out_size, void* d_ws, size_t ws_size,
                              hipStream_t stream) {
    (void)in_sizes; (void)n_in; (void)out_size; (void)ws_size;

    const float* z   = (const float*)d_in[0];
    const float* act = (const float*)d_in[1];
    const float* adj = (const float*)d_in[2];
    const float* w1  = (const float*)d_in[3];  const float* b1  = (const float*)d_in[4];
    const float* g1  = (const float*)d_in[5];  const float* be1 = (const float*)d_in[6];
    const float* w2  = (const float*)d_in[7];  const float* b2  = (const float*)d_in[8];
    const float* g2  = (const float*)d_in[9];  const float* be2 = (const float*)d_in[10];
    const float* wm  = (const float*)d_in[11]; const float* bm  = (const float*)d_in[12];
    const float* wv  = (const float*)d_in[13]; const float* bv  = (const float*)d_in[14];
    const float* wh1 = (const float*)d_in[15]; const float* bh1 = (const float*)d_in[16];
    const float* wh2 = (const float*)d_in[17]; const float* bh2 = (const float*)d_in[18];

    float* out = (float*)d_out;
    float* zState = out;                                   // final z (parity-0 buffer)
    float* maskP  = out + (size_t)kBatch * kDC;
    float* epsP   = out + 2 * (size_t)kBatch * kDC;
    int*   stepsOut = (int*)(out + 3 * (size_t)kBatch * kDC);

    // -------- workspace layout (with post-encoder aliasing) --------
    char* ws = (char*)d_ws;
    size_t o = 0;
    auto take = [&](size_t bytes) { size_t r = o; o += (bytes + 255) & ~(size_t)255; return r; };
    const size_t o_w1b  = take((size_t)kDA * kDH * 2);
    const size_t o_w2b  = take((size_t)kDH * kDH * 2);
    const size_t o_wmb  = take((size_t)kDH * kDC * 2);
    const size_t o_wvb  = take((size_t)kDH * kDC * 2);
    const size_t o_adjb = take((size_t)kDC * kDC * 2);
    const size_t o_wh1b = take((size_t)kKHalt * kDC * 2);
    const size_t o_actb = take((size_t)kBatch * kDA * 2);
    const size_t o_hb   = take((size_t)kBatch * kDH * 2);
    const size_t o_tmp  = take((size_t)kBatch * kDH * 4);   // fp32, encoder only
    const size_t o_h1b  = take((size_t)kBatch * kDH * 2);   // bf16, encoder only
    const size_t o_hpP  = take(2048 * 4);
    const size_t o_flag = take(16 * 4);
    const size_t o_step = take(4);

    bf16t* w1b  = (bf16t*)(ws + o_w1b);
    bf16t* w2b  = (bf16t*)(ws + o_w2b);
    bf16t* wmb  = (bf16t*)(ws + o_wmb);
    bf16t* wvb  = (bf16t*)(ws + o_wvb);
    bf16t* adjb = (bf16t*)(ws + o_adjb);
    bf16t* wh1b = (bf16t*)(ws + o_wh1b);
    bf16t* actb = (bf16t*)(ws + o_actb);
    bf16t* hb   = (bf16t*)(ws + o_hb);
    float* tmp  = (float*)(ws + o_tmp);
    bf16t* h1b  = (bf16t*)(ws + o_h1b);
    float* hpPart = (float*)(ws + o_hpP);
    int*   flags  = (int*)(ws + o_flag);
    int*   steps  = (int*)(ws + o_step);

    // post-encoder aliases (tmp region: 256 MB; h1b region: 128 MB)
    const size_t ZB = (size_t)kBatch * kDC;   // 16.7M elems
    float* values = (float*)(ws + o_h1b);                  // 64 MB
    float* baseB  = (float*)(ws + o_h1b + ZB * 4);         // 64 MB
    float* zBuf1  = (float*)(ws + o_tmp);                  // 64 MB (parity-1 z)
    bf16t* zbA    = (bf16t*)(ws + o_tmp + ZB * 4);         // 32 MB
    bf16t* zbB    = (bf16t*)(ws + o_tmp + ZB * 4 + ZB * 2);// 32 MB
    bf16t* zb0    = (bf16t*)(ws + o_tmp + ZB * 4 + 2 * ZB * 2); // 32 MB
    bf16t* deltab = (bf16t*)(ws + o_tmp + ZB * 4 + 3 * ZB * 2); // 32 MB

    auto cvt = [&](const float* s, bf16t* d, long long n) {
        long long n4 = n >> 2;
        k_f32_to_bf16<<<dim3((unsigned)((n4 + 255) / 256)), 256, 0, stream>>>(s, d, n4);
    };

    k_init_state<<<1, 32, 0, stream>>>(flags, steps);

    // bf16 operand preparation
    cvt(w1,  w1b,  (long long)kDA * kDH);
    cvt(w2,  w2b,  (long long)kDH * kDH);
    cvt(wm,  wmb,  (long long)kDH * kDC);
    cvt(wv,  wvb,  (long long)kDH * kDC);
    cvt(adj, adjb, (long long)kDC * kDC);
    cvt(wh1, wh1b, (long long)kKHalt * kDC);
    cvt(act, actb, (long long)kBatch * kDA);

    // ---- action encoder ----
    k_gemm_bias<<<dim3(kDH / 128, kBatch / 128), 256, 0, stream>>>(
        actb, kDA, w1b, kDH, kDA, kDH, b1, tmp, 0);
    k_ln_gelu_bf16<<<kBatch, 256, 0, stream>>>(tmp, g1, be1, h1b, kDH);
    k_gemm_bias<<<dim3(kDH / 128, kBatch / 128), 256, 0, stream>>>(
        h1b, kDH, w2b, kDH, kDH, kDH, b2, tmp, 0);
    k_ln_gelu_bf16<<<kBatch, 256, 0, stream>>>(tmp, g2, be2, hb, kDH);

    // ---- heads ----
    k_gemm_bias<<<dim3(kDC / 128, kBatch / 128), 256, 0, stream>>>(
        hb, kDH, wmb, kDC, kDH, kDC, bm, maskP, 1 /*sigmoid*/);
    k_gemm_bias<<<dim3(kDC / 128, kBatch / 128), 256, 0, stream>>>(
        hb, kDH, wvb, kDC, kDH, kDC, bv, values, 0);

    // ---- epsilon / z0 / base (tmp region free now) ----
    cvt(z, zb0, (long long)kBatch * kDC);
    k_gemm_eps<<<dim3(kDC / 128, kBatch / 128), 256, 0, stream>>>(
        zb0, adjb, z, maskP, values, epsP, baseB, zState, zbA);

    // ---- fixed-point loop with learned halting (6 iterations, device-gated) ----
    for (int s = 0; s < kMaxDepth; ++s) {
        const bool even = (s & 1) == 0;
        const float* zin  = even ? zState : zBuf1;
        float*       zout = even ? zBuf1  : zState;   // 6 iters -> final in zState
        const bf16t* zbi  = even ? zbA : zbB;
        bf16t*       zbo  = even ? zbB : zbA;

        k_gemm_iter<<<dim3(kDC / 128, kBatch / 128), 256, 0, stream>>>(
            zbi, adjb, zin, baseB, maskP, epsP, flags, s, zout, zbo, deltab);
        k_halter<<<kBatch / 32, 256, 0, stream>>>(
            deltab, wh1b, bh1, wh2, bh2, hpPart, s);
        k_halt_reduce<<<1, 256, 0, stream>>>(hpPart, kBatch / 32, flags, steps, s, stepsOut);
    }
}
